// Seq2Seq_85676007620739
// MI455X (gfx1250) — matrix operand order repaired and checked by reading, not compiled, not run on hardware
//
#include <hip/hip_runtime.h>
#include <stdint.h>

// ---------------- problem constants ----------------
#define H     1024
#define G4H   4096        // 4*H
#define BATCH 64
#define TENC  512
#define TDEC  512

// ---------------- launch shape ----------------
#define NWG      32       // persistent workgroups (small -> cheap grid barrier)
#define NTHREADS 256      // 8 waves (wave32)
// 32 WG * 8 waves = 256 waves; each wave owns a 16x64 slab of the 64x4096 GEMM.

typedef __bf16  v16bf __attribute__((ext_vector_type(16)));
typedef float   v8f   __attribute__((ext_vector_type(8)));

union BfVec { v16bf v; uint4 q[2]; };

__device__ __forceinline__ unsigned short f2bf(float f) {
  unsigned u = __float_as_uint(f);
  unsigned r = u + 0x7FFFu + ((u >> 16) & 1u);   // round-to-nearest-even
  return (unsigned short)(r >> 16);
}
__device__ __forceinline__ float bf2f(unsigned short h) {
  return __uint_as_float(((unsigned)h) << 16);
}
__device__ __forceinline__ float sigm(float x) { return 1.0f / (1.0f + __expf(-x)); }

// =====================================================================
// Weight pack kernel: fp32 [4096 x Ktot] (row = gate output j, col = k)
// -> bf16 WMMA B-tile layout:
//   dst[((nTile*KT + kTile)*32 + lane)*16 + e]
//   with  n = nTile*16 + (lane&15),  k = kTile*32 + (e>>3)*16 + (lane>>4)*8 + (e&7)
// (B matrix 32x16: lanes 0-15 hold K=0..7,16..23 of column n, lanes 16-31 hold
//  K=8..15,24..31 -> 32B contiguous per lane.)
// For Ktot=2048, cols [0,1024) come from Wa (Wih1), [1024,2048) from Wb (Whh1).
// =====================================================================
__global__ void pack_weights(const float* __restrict__ Wa,
                             const float* __restrict__ Wb,
                             unsigned short* __restrict__ dst,
                             int Ktot) {
  long long idx = (long long)blockIdx.x * blockDim.x + threadIdx.x;
  long long total = (long long)G4H * Ktot;
  if (idx >= total) return;
  int e        = (int)(idx & 15);
  long long r1 = idx >> 4;
  int lane     = (int)(r1 & 31);
  long long r2 = r1 >> 5;
  int KT       = Ktot >> 5;
  int kTile    = (int)(r2 % KT);
  int nTile    = (int)(r2 / KT);
  int n = nTile * 16 + (lane & 15);
  int k = kTile * 32 + ((e >> 3) << 4) + ((lane >> 4) << 3) + (e & 7);
  float v = (k < H) ? Wa[(long long)n * H + k]
                    : Wb[(long long)n * H + (k - H)];
  dst[idx] = f2bf(v);
}

// =====================================================================
// Grid-wide sense-reversing barrier (persistent kernel, NWG blocks)
// =====================================================================
__device__ __forceinline__ void grid_barrier(unsigned* cnt, unsigned* gen) {
  __syncthreads();
  if (threadIdx.x == 0) {
    __threadfence();
    volatile unsigned* vg = gen;
    unsigned g = *vg;
    if (atomicAdd(cnt, 1u) == NWG - 1) {
      *cnt = 0u;
      __threadfence();
      atomicAdd(gen, 1u);
    } else {
      while (*vg == g) { __builtin_amdgcn_s_sleep(2); }
    }
    __threadfence();
  }
  __syncthreads();
}

// =====================================================================
// GEMM phase: gates[64 x 4096] = hcat[64 x K] (bf16, row stride 2H) @ Wp
// Wp is pre-packed bf16 B-tiles, Ksteps = K/32 (even).
// Each wave: m = gw&3 (16 batch rows), nblk = gw>>2 (64 output cols).
// Hand software-pipelined (2-deep ping-pong, constant buffer indices) and
// K-loop kept rolled (#pragma unroll 1) so the compiler cannot fully unroll
// and spill streamed weight tiles to scratch (seen in round 1).
// =====================================================================
__device__ __forceinline__ void gemm_phase(const unsigned short* __restrict__ hcat,
                                           const unsigned short* __restrict__ Wp,
                                           float* __restrict__ gates,
                                           int Ksteps) {
  const int lane = threadIdx.x & 31;
  const int gw   = blockIdx.x * (NTHREADS / 32) + (threadIdx.x >> 5); // 0..255
  const int m    = gw & 3;
  const int nblk = gw >> 2;                                           // 0..63
  const int hi   = lane >> 4;
  const int rowA = m * 16 + (lane & 15);
  const unsigned short* aRow = hcat + (long long)rowA * (2 * H);
  // per-lane weight stream base: tiles for this wave's 4 N-tiles are at
  // tile index (nblk*4 + j)*Ksteps + ks; each tile is 32 lanes * 16 bf16.
  const unsigned short* wBase = Wp + ((long long)(nblk * 4) * Ksteps * 32 + lane) * 16;
  const long long tileStride = 32 * 16;                 // elems per k-step tile
  const long long nStride    = (long long)Ksteps * tileStride; // elems per N-tile

  v8f acc[4] = {};

  BfVec A0, A1, B0[4], B1[4];

  // ---- prologue: load k-step 0 ----
  A0.q[0] = *(const uint4*)(aRow + hi * 8);
  A0.q[1] = *(const uint4*)(aRow + hi * 8 + 16);
#pragma unroll
  for (int j = 0; j < 4; ++j) {
    const unsigned short* bp = wBase + j * nStride;
    B0[j].q[0] = *(const uint4*)(bp);
    B0[j].q[1] = *(const uint4*)(bp + 8);
  }

#pragma unroll 1
  for (int ks = 0; ks < Ksteps; ks += 2) {
    // ---- issue loads for ks+1 while computing ks ----
    {
      const int kn = ks + 1;
      A1.q[0] = *(const uint4*)(aRow + kn * 32 + hi * 8);
      A1.q[1] = *(const uint4*)(aRow + kn * 32 + hi * 8 + 16);
#pragma unroll
      for (int j = 0; j < 4; ++j) {
        const unsigned short* bp = wBase + j * nStride + kn * tileStride;
        __builtin_prefetch(bp + 2 * tileStride, 0, 1);   // warm ks+3
        B1[j].q[0] = *(const uint4*)(bp);
        B1[j].q[1] = *(const uint4*)(bp + 8);
      }
    }
#pragma unroll
    for (int j = 0; j < 4; ++j)
      acc[j] = __builtin_amdgcn_wmma_f32_16x16x32_bf16(
          false, A0.v, false, B0[j].v, (short)0, acc[j], false, false);

    // ---- issue loads for ks+2 while computing ks+1 ----
    if (ks + 2 < Ksteps) {
      const int kn = ks + 2;
      A0.q[0] = *(const uint4*)(aRow + kn * 32 + hi * 8);
      A0.q[1] = *(const uint4*)(aRow + kn * 32 + hi * 8 + 16);
#pragma unroll
      for (int j = 0; j < 4; ++j) {
        const unsigned short* bp = wBase + j * nStride + kn * tileStride;
        B0[j].q[0] = *(const uint4*)(bp);
        B0[j].q[1] = *(const uint4*)(bp + 8);
      }
    }
#pragma unroll
    for (int j = 0; j < 4; ++j)
      acc[j] = __builtin_amdgcn_wmma_f32_16x16x32_bf16(
          false, A1.v, false, B1[j].v, (short)0, acc[j], false, false);
  }

  // C/D layout: lane = N col (0..15 | 16..31 -> +8 rows), VGPR r -> row m*16+r(+8)
#pragma unroll
  for (int j = 0; j < 4; ++j) {
    const int n = (nblk * 4 + j) * 16 + (lane & 15);
#pragma unroll
    for (int r = 0; r < 8; ++r) {
      const int mr = m * 16 + hi * 8 + r;
      gates[(long long)mr * G4H + n] = acc[j][r];
    }
  }
}

// =====================================================================
// Elementwise LSTM cell phase: gates (pre-activation, no bias) -> h,c.
// wih0 != null: add bias + x * Wih0 column (rank-1 input, input_size==1),
// x for batch b read from xp[b*xstride].
// Writes h as bf16 into hdst (the hcat A-staging buffer, row stride 2H).
// =====================================================================
__device__ __forceinline__ void elem_layer(const float* __restrict__ gates,
                                           const float* __restrict__ bias,
                                           const float* __restrict__ wih0,
                                           const float* __restrict__ xp, int xstride,
                                           float* __restrict__ c,
                                           unsigned short* __restrict__ hdst) {
  const int u0 = blockIdx.x * NTHREADS + threadIdx.x;   // 0..8191
#pragma unroll
  for (int it = 0; it < (BATCH * H) / (NWG * NTHREADS); ++it) {  // 8 iters
    const int idx = u0 + it * (NWG * NTHREADS);
    const int b = idx >> 10;
    const int j = idx & (H - 1);
    const float* g = gates + (long long)b * G4H;
    float gi = g[j]         + bias[j];
    float gf = g[H + j]     + bias[H + j];
    float gg = g[2 * H + j] + bias[2 * H + j];
    float go = g[3 * H + j] + bias[3 * H + j];
    if (wih0) {
      const float x = xp[b * xstride];
      gi += x * wih0[j];
      gf += x * wih0[H + j];
      gg += x * wih0[2 * H + j];
      go += x * wih0[3 * H + j];
    }
    const float cp = c[b * H + j];
    const float cn = sigm(gf) * cp + sigm(gi) * tanhf(gg);
    const float hn = sigm(go) * tanhf(cn);
    c[b * H + j] = cn;
    hdst[(long long)b * (2 * H) + j] = f2bf(hn);
  }
}

// =====================================================================
// Persistent seq2seq kernel: all 512 enc + 512 dec steps, grid-synced.
// =====================================================================
__global__ void __launch_bounds__(NTHREADS, 1)
seq2seq_kernel(const float* __restrict__ prev, const float* __restrict__ nxt,
               const unsigned char* __restrict__ tf_mask,
               const float* __restrict__ eWih0, const float* __restrict__ eb0,
               const float* __restrict__ eb1,
               const float* __restrict__ dWih0, const float* __restrict__ db0,
               const float* __restrict__ db1,
               const float* __restrict__ fcW, const float* __restrict__ fcb,
               const unsigned short* __restrict__ Pe0,
               const unsigned short* __restrict__ Pe1,
               const unsigned short* __restrict__ Pd0,
               const unsigned short* __restrict__ Pd1,
               float* __restrict__ gates, unsigned short* __restrict__ hcat,
               float* __restrict__ c0, float* __restrict__ c1,
               float* __restrict__ xbuf, unsigned* __restrict__ bar,
               float* __restrict__ out) {
  unsigned* bcnt = bar;
  unsigned* bgen = bar + 1;

  // ---------------- encoder ----------------
  for (int t = 0; t < TENC; ++t) {
    gemm_phase(hcat, Pe0, gates, H / 32);          // h0 @ Whh0^T
    grid_barrier(bcnt, bgen);
    elem_layer(gates, eb0, eWih0, prev + t, TENC, c0, hcat);        // -> h0
    grid_barrier(bcnt, bgen);
    gemm_phase(hcat, Pe1, gates, (2 * H) / 32);    // [h0,h1] @ [Wih1;Whh1]^T
    grid_barrier(bcnt, bgen);
    elem_layer(gates, eb1, nullptr, nullptr, 0, c1, hcat + H);      // -> h1
    grid_barrier(bcnt, bgen);
  }

  // ---------------- decoder (teacher forcing) ----------------
  for (int t = 0; t < TDEC; ++t) {
    gemm_phase(hcat, Pd0, gates, H / 32);
    grid_barrier(bcnt, bgen);
    elem_layer(gates, db0, dWih0, xbuf, 1, c0, hcat);
    grid_barrier(bcnt, bgen);
    gemm_phase(hcat, Pd1, gates, (2 * H) / 32);
    grid_barrier(bcnt, bgen);
    elem_layer(gates, db1, nullptr, nullptr, 0, c1, hcat + H);
    grid_barrier(bcnt, bgen);
    // fc1 + teacher-forcing select: 64 dot products of length H
    if (blockIdx.x == 0 && threadIdx.x < BATCH) {
      const int b = threadIdx.x;
      const unsigned short* h1 = hcat + (long long)b * (2 * H) + H;
      float s = fcb[0];
      for (int k = 0; k < H; ++k) s += bf2f(h1[k]) * fcW[k];
      out[b * TDEC + t] = s;
      xbuf[b] = tf_mask[t] ? nxt[b * TDEC + t] : s;
    }
    grid_barrier(bcnt, bgen);
  }
}

// =====================================================================
// Host launcher
// =====================================================================
extern "C" void kernel_launch(void* const* d_in, const int* in_sizes, int n_in,
                              void* d_out, int out_size, void* d_ws, size_t ws_size,
                              hipStream_t stream) {
  (void)in_sizes; (void)n_in; (void)out_size; (void)ws_size;
  const float* prev = (const float*)d_in[0];
  const float* nxt  = (const float*)d_in[1];
  const unsigned char* tfm = (const unsigned char*)d_in[2]; // jnp bool_ (1 byte)
  const float* eWih0 = (const float*)d_in[3];
  const float* eWhh0 = (const float*)d_in[4];
  const float* eb0   = (const float*)d_in[5];
  const float* eWih1 = (const float*)d_in[6];
  const float* eWhh1 = (const float*)d_in[7];
  const float* eb1   = (const float*)d_in[8];
  const float* dWih0 = (const float*)d_in[9];
  const float* dWhh0 = (const float*)d_in[10];
  const float* db0   = (const float*)d_in[11];
  const float* dWih1 = (const float*)d_in[12];
  const float* dWhh1 = (const float*)d_in[13];
  const float* db1   = (const float*)d_in[14];
  const float* fcW   = (const float*)d_in[15];
  const float* fcb   = (const float*)d_in[16];

  char* ws = (char*)d_ws;
  const size_t SZ_P0 = (size_t)G4H * H * 2;         //  8 MB (K=1024 packed bf16)
  const size_t SZ_P1 = (size_t)G4H * 2 * H * 2;     // 16 MB (K=2048 packed bf16)
  unsigned short* Pe0 = (unsigned short*)(ws);
  unsigned short* Pe1 = (unsigned short*)(ws + SZ_P0);
  unsigned short* Pd0 = (unsigned short*)(ws + SZ_P0 + SZ_P1);
  unsigned short* Pd1 = (unsigned short*)(ws + 2 * SZ_P0 + SZ_P1);
  char* dyn = ws + 2 * SZ_P0 + 2 * SZ_P1;           // ~48 MB mark

  float*          gates = (float*)dyn;                                   // 1 MB
  unsigned short* hcat  = (unsigned short*)(dyn + (size_t)BATCH * G4H * 4);
  float*          c0    = (float*)((char*)hcat + (size_t)BATCH * 2 * H * 2);
  float*          c1    = c0 + (size_t)BATCH * H;
  float*          xbuf  = c1 + (size_t)BATCH * H;
  unsigned*       bar   = (unsigned*)(xbuf + BATCH);
  const size_t dynBytes = (size_t)BATCH * G4H * 4       // gates
                        + (size_t)BATCH * 2 * H * 2     // hcat (h0|h1 bf16)
                        + 2 * (size_t)BATCH * H * 4     // c0, c1
                        + BATCH * 4 + 64;               // xbuf + barrier words

  // zero state + barrier each call (graph-capture safe, deterministic)
  hipMemsetAsync(dyn, 0, dynBytes, stream);

  const int tp = 256;
  const long long tot0 = (long long)G4H * H;
  const long long tot1 = (long long)G4H * 2 * H;
  pack_weights<<<(unsigned)((tot0 + tp - 1) / tp), tp, 0, stream>>>(eWhh0, nullptr, Pe0, H);
  pack_weights<<<(unsigned)((tot1 + tp - 1) / tp), tp, 0, stream>>>(eWih1, eWhh1,  Pe1, 2 * H);
  pack_weights<<<(unsigned)((tot0 + tp - 1) / tp), tp, 0, stream>>>(dWhh0, nullptr, Pd0, H);
  pack_weights<<<(unsigned)((tot1 + tp - 1) / tp), tp, 0, stream>>>(dWih1, dWhh1,  Pd1, 2 * H);

  seq2seq_kernel<<<NWG, NTHREADS, 0, stream>>>(
      prev, nxt, tfm, eWih0, eb0, eb1, dWih0, db0, db1, fcW, fcb,
      Pe0, Pe1, Pd0, Pd1, gates, hcat, c0, c1, xbuf, bar, (float*)d_out);
}
